// SPMultiHeadedAttention_14731737825404
// MI455X (gfx1250) — matrix operand order repaired and checked by reading, not compiled, not run on hardware
//
#include <hip/hip_runtime.h>

// ---------------------------------------------------------------------------
// CDNA5 (gfx1250) fused pre-LN multi-head attention block.
// GEMMs: v_wmma_f32_16x16x32_bf16 fed from LDS, tiles DMA'd by the Tensor
// Data Mover (tensor_load_to_lds, TENSORcnt), double-buffered.
// ---------------------------------------------------------------------------

typedef __attribute__((ext_vector_type(16))) __bf16 v16bf;
typedef __attribute__((ext_vector_type(8)))  float  v8f;
typedef __attribute__((ext_vector_type(4)))  unsigned int tdm_v4u;
typedef __attribute__((ext_vector_type(8)))  int          tdm_v8i;
typedef __attribute__((ext_vector_type(4)))  int          tdm_v4i;

union WmmaOp {
  uint4 u[2];   // 32 bytes
  v16bf v;
};

__device__ __forceinline__ unsigned short f2bf(float f) {
  // round-to-nearest-even fp32 -> bf16
  unsigned u = __float_as_uint(f);
  u += 0x7fffu + ((u >> 16) & 1u);
  return (unsigned short)(u >> 16);
}

__device__ __forceinline__ v8f wmma_bf16(const WmmaOp& a, const WmmaOp& b, v8f c) {
  return __builtin_amdgcn_wmma_f32_16x16x32_bf16(false, a.v, false, b.v,
                                                 (short)0, c, false, false);
}

// ---------------------------------------------------------------------------
// Problem constants
// ---------------------------------------------------------------------------
#define BATCH   2048
#define SEQ     28
#define DMODEL  1024
#define NHEAD   16
#define DHEAD   64
#define MROWS   (BATCH * SEQ)          // 57344
#define NQKV    (3 * DMODEL)           // 3072

// ---------------------------------------------------------------------------
// TDM: DMA a [rows x 32] bf16 tile (row stride = strideElems) from global
// memory into LDS, padding each 64B row with 32B so the LDS row stride is
// 48 bf16 elements (= LDSS below).  D# layout per CDNA5 ISA §8.3/§8.4.
// ---------------------------------------------------------------------------
__device__ __forceinline__ void tdm_load_tile(const unsigned short* gsrc,
                                              unsigned lds_byte_addr,
                                              int strideElems, int K, int rows) {
  unsigned long long ga = (unsigned long long)(uintptr_t)gsrc;
  tdm_v4u g0;
  g0[0] = 1u;                                          // count=1, user mode
  g0[1] = lds_byte_addr;                               // LDS byte address
  g0[2] = (unsigned)ga;                                // global_addr[31:0]
  g0[3] = (unsigned)((ga >> 32) & 0x1FFFFFFu)          // global_addr[56:32]
        | (2u << 30);                                  // type = 2 (image)
  tdm_v8i g1;
  // d0: workgroup_mask=0 | data_size=1(2B)<<16 | pad_enable<<20 |
  //     pad_interval=3(16 DW)<<22 | pad_amount=7(8 DW)<<25
  g1[0] = (int)((1u << 16) | (1u << 20) | (3u << 22) | (7u << 25));
  g1[1] = (int)((unsigned)(K & 0xFFFF) << 16);         // tensor_dim0 lo16
  g1[2] = (int)(((unsigned)(K >> 16) & 0xFFFFu)        // tensor_dim0 hi16
        | (16384u << 16));                             // tensor_dim1 lo16 (big)
  g1[3] = (int)(32u << 16);                            // dim1 hi=0 | tile_dim0=32
  g1[4] = (int)(unsigned)rows;                         // tile_dim1=rows, tile_dim2=0
  g1[5] = strideElems;                                 // tensor_dim0_stride lo32
  g1[6] = 0;                                           // stride hi | dim1_stride lo
  g1[7] = 0;
  tdm_v4i z4 = (tdm_v4i)0;
#if defined(__clang_major__) && __clang_major__ >= 23
  tdm_v8i z8 = (tdm_v8i)0;
  __builtin_amdgcn_tensor_load_to_lds(g0, g1, z4, z4, z8, 0);
#else
  __builtin_amdgcn_tensor_load_to_lds(g0, g1, z4, z4, 0);
#endif
}

__device__ __forceinline__ unsigned lds_addr_of(const void* p) {
  // flat address of LDS = {shared aperture, lds offset}; low 32 bits is the
  // byte address the TDM descriptor wants.
  return (unsigned)(uintptr_t)p;
}

// ---------------------------------------------------------------------------
// Kernel 1: pack weights -> bf16, transposed to N-major (B operand wants
// lane = N column holding contiguous K values).
// ---------------------------------------------------------------------------
__global__ __launch_bounds__(256)
void prep_weights_kernel(const float* __restrict__ wq, const float* __restrict__ wk,
                         const float* __restrict__ wv, const float* __restrict__ wo,
                         unsigned short* __restrict__ wqkv_t,
                         unsigned short* __restrict__ wo_t) {
  int idx = blockIdx.x * 256 + threadIdx.x;
  const int QKV = NQKV * DMODEL;       // 3,145,728
  if (idx < QKV) {
    int n = idx >> 10, k = idx & 1023;
    int sel = n >> 10, col = n & 1023;
    const float* w = (sel == 0) ? wq : (sel == 1) ? wk : wv;
    wqkv_t[idx] = f2bf(w[k * DMODEL + col]);
  } else {
    int j = idx - QKV;                 // < 1024*1024
    int n = j >> 10, k = j & 1023;
    wo_t[j] = f2bf(wo[k * DMODEL + n]);
  }
}

// ---------------------------------------------------------------------------
// Kernel 2: LayerNorm.  One 256-thread block per row of 1024.
// ---------------------------------------------------------------------------
__global__ __launch_bounds__(256)
void layernorm_kernel(const float* __restrict__ x,
                      const float* __restrict__ gamma,
                      const float* __restrict__ beta,
                      float* __restrict__ hout,
                      unsigned short* __restrict__ h16) {
  __shared__ float red[16];
  const int row = blockIdx.x;
  const int tid = threadIdx.x;
  const float* xr = x + (size_t)row * DMODEL;

  float4 v = *(const float4*)(xr + tid * 4);
  float s  = v.x + v.y + v.z + v.w;
  float sq = v.x * v.x + v.y * v.y + v.z * v.z + v.w * v.w;
  #pragma unroll
  for (int off = 16; off; off >>= 1) {
    s  += __shfl_xor(s,  off, 32);
    sq += __shfl_xor(sq, off, 32);
  }
  const int wv32 = tid >> 5, lane = tid & 31;
  if (lane == 0) { red[wv32] = s; red[8 + wv32] = sq; }
  __syncthreads();
  float ts = 0.f, tsq = 0.f;
  #pragma unroll
  for (int w = 0; w < 8; ++w) { ts += red[w]; tsq += red[8 + w]; }
  const float mean = ts * (1.0f / DMODEL);
  const float var  = tsq * (1.0f / DMODEL) - mean * mean;
  const float rstd = rsqrtf(var + 1e-5f);

  float4 g  = *(const float4*)(gamma + tid * 4);
  float4 be = *(const float4*)(beta  + tid * 4);
  float4 y;
  y.x = (v.x - mean) * rstd * g.x + be.x;
  y.y = (v.y - mean) * rstd * g.y + be.y;
  y.z = (v.z - mean) * rstd * g.z + be.z;
  y.w = (v.w - mean) * rstd * g.w + be.w;

  *(float4*)(hout + (size_t)row * DMODEL + tid * 4) = y;
  ushort4 yb;
  yb.x = f2bf(y.x); yb.y = f2bf(y.y); yb.z = f2bf(y.z); yb.w = f2bf(y.w);
  *(ushort4*)(h16 + (size_t)row * DMODEL + tid * 4) = yb;
}

// ---------------------------------------------------------------------------
// Kernel 3/5: bf16 GEMM  C[M,N] = A[M,K] @ Bt[N,K]^T, fp32 accumulate.
// 256 threads = 8 waves; tile 128x128; wave tile 64x32 (4x2 WMMAs).
// K stepped by 32.  Tiles DMA'd global->LDS by the TDM (wave 0 issues,
// TENSORcnt-waited), double-buffered so the DMA overlaps the 8 WMMAs.
// mode 0: C -> bf16 out16.   mode 1: outf[idx] += C (residual add).
// ---------------------------------------------------------------------------
#define BM 128
#define BN 128
#define BK 32
#define LDSS 48   // padded LDS row stride in bf16 elements (96B, 16B-aligned)

__global__ __launch_bounds__(256)
void gemm_bf16_kernel(const unsigned short* __restrict__ A,
                      const unsigned short* __restrict__ Bt,
                      unsigned short* __restrict__ out16,
                      float* __restrict__ outf,
                      int M, int N, int K, int mode) {
  __shared__ unsigned short As[2][BM * LDSS];
  __shared__ unsigned short Bs[2][BN * LDSS];

  const int tid  = threadIdx.x;
  const int lane = tid & 31;
  const int wave = tid >> 5;
  const int bm = blockIdx.y * BM;
  const int bn = blockIdx.x * BN;
  const int wm = (wave >> 2) * 64;     // 2 wave-rows
  const int wn = (wave & 3) * 32;      // 4 wave-cols
  const int half = lane >> 4;          // 0: lanes 0-15, 1: lanes 16-31
  const int l15  = lane & 15;
  const int koA  = half * 8;           // A operand: K chunks koA..+7, koA+16..+23
  const int koB  = half * 8;           // B operand: K chunks koB..+7, koB+16..+23

  v8f acc[4][2];
  #pragma unroll
  for (int i = 0; i < 4; ++i)
    #pragma unroll
    for (int j = 0; j < 2; ++j) acc[i][j] = (v8f){};

  const unsigned short* aTile = A  + (size_t)bm * K;
  const unsigned short* bTile = Bt + (size_t)bn * K;

  // Prologue: DMA tile 0 into buffer 0 (both matrices), wave 0 only.
  if (wave == 0) {
    tdm_load_tile(aTile, lds_addr_of(&As[0][0]), K, K, BM);
    tdm_load_tile(bTile, lds_addr_of(&Bs[0][0]), K, K, BN);
  }

  int cur = 0;
  for (int k0 = 0; k0 < K; k0 += BK) {
    if (wave == 0) __builtin_amdgcn_s_wait_tensorcnt(0);  // tile `cur` landed
    __syncthreads();                                      // ...for all waves
    if (wave == 0 && (k0 + BK) < K) {                     // DMA next tile,
      tdm_load_tile(aTile + k0 + BK, lds_addr_of(&As[cur ^ 1][0]), K, K, BM);
      tdm_load_tile(bTile + k0 + BK, lds_addr_of(&Bs[cur ^ 1][0]), K, K, BN);
    }                                                     // overlaps WMMAs

    const unsigned short* as = As[cur];
    const unsigned short* bs = Bs[cur];
    WmmaOp a[4], b[2];
    #pragma unroll
    for (int mi = 0; mi < 4; ++mi) {
      const unsigned short* p = &as[(wm + mi * 16 + l15) * LDSS];
      a[mi].u[0] = *(const uint4*)(p + koA);
      a[mi].u[1] = *(const uint4*)(p + koA + 16);
    }
    #pragma unroll
    for (int ni = 0; ni < 2; ++ni) {
      const unsigned short* p = &bs[(wn + ni * 16 + l15) * LDSS];
      b[ni].u[0] = *(const uint4*)(p + koB);
      b[ni].u[1] = *(const uint4*)(p + koB + 16);
    }
    #pragma unroll
    for (int mi = 0; mi < 4; ++mi)
      #pragma unroll
      for (int ni = 0; ni < 2; ++ni)
        acc[mi][ni] = wmma_bf16(a[mi], b[ni], acc[mi][ni]);
    cur ^= 1;
  }

  // Epilogue.  C layout: VGPR r -> M = r + 8*half (within 16-tile), N = l15.
  #pragma unroll
  for (int mi = 0; mi < 4; ++mi) {
    #pragma unroll
    for (int ni = 0; ni < 2; ++ni) {
      #pragma unroll
      for (int r = 0; r < 8; ++r) {
        const int grow = bm + wm + mi * 16 + r + half * 8;
        const int gcol = bn + wn + ni * 16 + l15;
        const size_t idx = (size_t)grow * N + gcol;
        const float vv = acc[mi][ni][r];
        if (mode == 0) out16[idx] = f2bf(vv);
        else           outf[idx] += vv;
      }
    }
  }
}

// ---------------------------------------------------------------------------
// Kernel 4: fused attention, one wave32 per (batch, head).
// scores(28x28, padded 32x32) = Q K^T via 8 WMMAs; +teeth-ALiBi bias; masked
// softmax via 16-lane shuffles; P re-laid-out through LDS; V transposed
// through LDS; ctx = P V via 8 WMMAs.  qkv rows: [q(1024) k(1024) v(1024)].
// ---------------------------------------------------------------------------
__global__ __launch_bounds__(32)
void attention_kernel(const unsigned short* __restrict__ qkv,
                      unsigned short* __restrict__ ctx) {
  __shared__ unsigned short P [32 * 32];   // probs, bf16
  __shared__ unsigned short Vt[64 * 32];   // V^T: [d][j], bf16

  const int lane = threadIdx.x;
  const int b = blockIdx.x >> 4;
  const int h = blockIdx.x & 15;
  const unsigned short* qb = qkv + (size_t)b * SEQ * NQKV + h * DHEAD;
  const unsigned short* kb = qb + DMODEL;
  const unsigned short* vb = qb + 2 * DMODEL;

  // ---- stage V^T into LDS (zero-fill padded rows j=28..31) -------------
  if (lane < SEQ) {
    const unsigned short* vr = vb + (size_t)lane * NQKV;
    #pragma unroll
    for (int d = 0; d < DHEAD; ++d) Vt[d * 32 + lane] = vr[d];
  } else {
    #pragma unroll
    for (int d = 0; d < DHEAD; ++d) Vt[d * 32 + lane] = 0;
  }

  const int half = lane >> 4;
  const int l15  = lane & 15;
  const int koA  = half * 8;
  const int koB  = half * 8;

  // ---- scores = Q @ K^T  (2x2 tiles of 16x16, K=64 in two 32-steps) ----
  v8f s[2][2];
  #pragma unroll
  for (int i = 0; i < 2; ++i)
    #pragma unroll
    for (int j = 0; j < 2; ++j) s[i][j] = (v8f){};

  #pragma unroll
  for (int ks = 0; ks < 2; ++ks) {
    WmmaOp aq[2], bk[2];
    #pragma unroll
    for (int it = 0; it < 2; ++it) {
      int row = it * 16 + l15; if (row > SEQ - 1) row = SEQ - 1;   // clamp pad
      const unsigned short* p = qb + (size_t)row * NQKV + ks * 32;
      aq[it].u[0] = *(const uint4*)(p + koA);
      aq[it].u[1] = *(const uint4*)(p + koA + 16);
    }
    #pragma unroll
    for (int jt = 0; jt < 2; ++jt) {
      int row = jt * 16 + l15; if (row > SEQ - 1) row = SEQ - 1;
      const unsigned short* p = kb + (size_t)row * NQKV + ks * 32;
      bk[jt].u[0] = *(const uint4*)(p + koB);
      bk[jt].u[1] = *(const uint4*)(p + koB + 16);
    }
    #pragma unroll
    for (int it = 0; it < 2; ++it)
      #pragma unroll
      for (int jt = 0; jt < 2; ++jt)
        s[it][jt] = wmma_bf16(aq[it], bk[jt], s[it][jt]);
  }

  // ---- bias + mask + softmax ------------------------------------------
  const float slope = exp2f(-0.5f * (float)(h + 1));
  const int j0 = l15, j1 = 16 + l15;
  #pragma unroll
  for (int it = 0; it < 2; ++it) {
    #pragma unroll
    for (int r = 0; r < 8; ++r) {
      const int i = it * 16 + r + half * 8;        // this lane-half's row
      float v0 = s[it][0][r], v1 = s[it][1][r];
      if (i < SEQ && j0 < SEQ) {
        int bias = abs(i % 14 - j0 % 14) + abs(i / 14 - j0 / 14);
        v0 += slope * (float)bias;
      } else v0 = -1e30f;
      if (i < SEQ && j1 < SEQ) {
        int bias = abs(i % 14 - j1 % 14) + abs(i / 14 - j1 / 14);
        v1 += slope * (float)bias;
      } else v1 = -1e30f;

      float m = fmaxf(v0, v1);                     // row max over 16 lanes
      #pragma unroll
      for (int off = 8; off; off >>= 1) m = fmaxf(m, __shfl_xor(m, off, 32));
      float e0 = __expf(v0 - m), e1 = __expf(v1 - m);
      float sum = e0 + e1;
      #pragma unroll
      for (int off = 8; off; off >>= 1) sum += __shfl_xor(sum, off, 32);
      const float inv = 1.0f / sum;
      P[i * 32 + j0] = f2bf(e0 * inv);
      P[i * 32 + j1] = f2bf(e1 * inv);
    }
  }
  __syncthreads();

  // ---- ctx = P @ V  (K = 32 covers all j in one WMMA) ------------------
  #pragma unroll
  for (int it = 0; it < 2; ++it) {
    WmmaOp ap;
    const unsigned short* pp = &P[(it * 16 + l15) * 32];
    ap.u[0] = *(const uint4*)(pp + koA);
    ap.u[1] = *(const uint4*)(pp + koA + 16);
    #pragma unroll
    for (int nt = 0; nt < 4; ++nt) {
      WmmaOp bv;
      const unsigned short* vp = &Vt[(nt * 16 + l15) * 32];
      bv.u[0] = *(const uint4*)(vp + koB);
      bv.u[1] = *(const uint4*)(vp + koB + 16);
      v8f c = (v8f){};
      c = wmma_bf16(ap, bv, c);
      #pragma unroll
      for (int r = 0; r < 8; ++r) {
        const int i = it * 16 + r + half * 8;
        if (i < SEQ) {
          const size_t o = (size_t)(b * SEQ + i) * DMODEL
                         + h * DHEAD + nt * 16 + l15;
          ctx[o] = f2bf(c[r]);
        }
      }
    }
  }
}

// ---------------------------------------------------------------------------
// Host launcher.  Workspace layout (bytes, all 256-aligned):
//   h16     [57344*1024] bf16 : 117,440,512
//   wqkv_t  [3072*1024]  bf16 :   6,291,456
//   wo_t    [1024*1024]  bf16 :   2,097,152
//   qkv     [57344*3072] bf16 : 352,321,536
//   ctx     [57344*1024] bf16 : 117,440,512     total ~595.6 MB
// ---------------------------------------------------------------------------
extern "C" void kernel_launch(void* const* d_in, const int* in_sizes, int n_in,
                              void* d_out, int out_size, void* d_ws, size_t ws_size,
                              hipStream_t stream) {
  const float* hidden = (const float*)d_in[0];
  const float* wq     = (const float*)d_in[1];
  const float* wk     = (const float*)d_in[2];
  const float* wv     = (const float*)d_in[3];
  const float* wo     = (const float*)d_in[4];
  const float* gamma  = (const float*)d_in[5];
  const float* beta   = (const float*)d_in[6];
  float* out = (float*)d_out;

  char* ws = (char*)d_ws;
  unsigned short* h16    = (unsigned short*)(ws);
  unsigned short* wqkv_t = (unsigned short*)(ws + 117440512ull);
  unsigned short* wo_t   = (unsigned short*)(ws + 123731968ull);
  unsigned short* qkv    = (unsigned short*)(ws + 125829120ull);
  unsigned short* ctx    = (unsigned short*)(ws + 478150656ull);

  // 1) weights -> bf16 transposed
  prep_weights_kernel<<<(NQKV * DMODEL + DMODEL * DMODEL) / 256, 256, 0, stream>>>(
      wq, wk, wv, wo, wqkv_t, wo_t);

  // 2) LayerNorm: d_out = h (residual base), h16 = bf16(h)
  layernorm_kernel<<<MROWS, 256, 0, stream>>>(hidden, gamma, beta, out, h16);

  // 3) qkv = h16 @ wqkv_t^T   [57344 x 3072]
  gemm_bf16_kernel<<<dim3(NQKV / BN, MROWS / BM), 256, 0, stream>>>(
      h16, wqkv_t, qkv, nullptr, MROWS, NQKV, DMODEL, 0);

  // 4) fused attention -> ctx [57344 x 1024]
  attention_kernel<<<BATCH * NHEAD, 32, 0, stream>>>(qkv, ctx);

  // 5) d_out += ctx @ wo_t^T  (residual add onto h)
  gemm_bf16_kernel<<<dim3(DMODEL / BN, MROWS / BM), 256, 0, stream>>>(
      ctx, wo_t, nullptr, out, MROWS, DMODEL, DMODEL, 1);
}
